// QLinearTile2D_BSR_30975304139139
// MI455X (gfx1250) — matrix operand order repaired and checked by reading, not compiled, not run on hardware
//
#include <hip/hip_runtime.h>

// Problem constants (from the reference)
#define R_BLOCKS 32
#define K_BLOCKS 8
#define TS       128
#define C_IN     4096
#define N_ROWS   4096      // B*T = 2*2048
#define C_OUT    4096      // R*TS
#define KC       32        // K-depth per WMMA step
#define M_TILE   128       // rows per workgroup
#define LDS_STRIDE 40      // 32 data + 8 pad (ushorts) -> conflict-free b128 reads

#define XN  ((size_t)N_ROWS * C_IN)                        // 16,777,216 elems
#define WN  ((size_t)R_BLOCKS * K_BLOCKS * TS * TS)        // 4,194,304 elems
#define NCHUNK (K_BLOCKS * (TS / KC))                      // 32

typedef __attribute__((ext_vector_type(16))) __bf16 v16bf;
typedef __attribute__((ext_vector_type(8)))  float  v8f;

union Frag {
  v16bf v;
  uint4 q[2];
};

// Split fp32 into bf16 hi + bf16 lo (round-to-nearest-even), x ~= hi + lo.
__device__ inline void splitBf16(float x, unsigned short& hi, unsigned short& lo) {
  unsigned int u  = __float_as_uint(x);
  unsigned int rh = u + 0x7FFFu + ((u >> 16) & 1u);
  hi = (unsigned short)(rh >> 16);
  float fhi = __uint_as_float(((unsigned int)hi) << 16);
  float res = x - fhi;
  unsigned int v  = __float_as_uint(res);
  unsigned int rl = v + 0x7FFFu + ((v >> 16) & 1u);
  lo = (unsigned short)(rl >> 16);
}

__device__ inline unsigned int pk2(unsigned short a, unsigned short b) {
  return (unsigned int)a | ((unsigned int)b << 16);
}

// CDNA5 async global->LDS copy (ASYNCcnt-tracked). vdst = per-lane LDS byte
// address, vaddr = per-lane 64-bit global address.
__device__ inline void asyncCopyB128(unsigned lds_byte_off, const void* g) {
  asm volatile("global_load_async_to_lds_b128 %0, %1, off"
               :: "v"(lds_byte_off),
                  "v"((unsigned long long)(size_t)g)
               : "memory");
}

#define WAIT_ASYNCCNT_4() asm volatile("s_wait_asynccnt 0x4" ::: "memory")
#define WAIT_ASYNCCNT_0() asm volatile("s_wait_asynccnt 0x0" ::: "memory")

// ---------------------------------------------------------------------------
// Pass 1: one-shot fp32 -> (bf16_hi, bf16_lo) conversion for x and blocks.
// ---------------------------------------------------------------------------
__global__ __launch_bounds__(256)
void convert_split_kernel(const float* __restrict__ x,
                          const float* __restrict__ blocks,
                          unsigned short* __restrict__ xhi,
                          unsigned short* __restrict__ xlo,
                          unsigned short* __restrict__ whi,
                          unsigned short* __restrict__ wlo)
{
  size_t g = (size_t)blockIdx.x * blockDim.x + threadIdx.x;  // 8-element group
  const float* src;
  unsigned short *dhi, *dlo;
  size_t base;
  if (g < XN / 8) {
    src = x;      dhi = xhi; dlo = xlo; base = g * 8;
  } else {
    src = blocks; dhi = whi; dlo = wlo; base = (g - XN / 8) * 8;
  }
  float4 a = ((const float4*)(src + base))[0];
  float4 b = ((const float4*)(src + base))[1];
  unsigned short h[8], l[8];
  splitBf16(a.x, h[0], l[0]); splitBf16(a.y, h[1], l[1]);
  splitBf16(a.z, h[2], l[2]); splitBf16(a.w, h[3], l[3]);
  splitBf16(b.x, h[4], l[4]); splitBf16(b.y, h[5], l[5]);
  splitBf16(b.z, h[6], l[6]); splitBf16(b.w, h[7], l[7]);
  *(uint4*)(dhi + base) = make_uint4(pk2(h[0], h[1]), pk2(h[2], h[3]),
                                     pk2(h[4], h[5]), pk2(h[6], h[7]));
  *(uint4*)(dlo + base) = make_uint4(pk2(l[0], l[1]), pk2(l[2], l[3]),
                                     pk2(l[4], l[5]), pk2(l[6], l[7]));
}

// ---------------------------------------------------------------------------
// Pass 2: WMMA GEMM. A fragments: direct per-lane global b128 loads from the
// pre-split x planes. W chunks: async global->LDS (double-buffered), shared by
// all 8 waves; B fragments read from LDS with conflict-free padding.
// ---------------------------------------------------------------------------
__global__ __launch_bounds__(256)
void bsr_wmma_ws_kernel(const unsigned short* __restrict__ xhi,
                        const unsigned short* __restrict__ xlo,
                        const unsigned short* __restrict__ whi,
                        const unsigned short* __restrict__ wlo,
                        const int* __restrict__ col_idx,
                        float* __restrict__ y)
{
  __shared__ __attribute__((aligned(16))) unsigned short sWhi[2][TS * LDS_STRIDE];
  __shared__ __attribute__((aligned(16))) unsigned short sWlo[2][TS * LDS_STRIDE];

  const int r    = blockIdx.y;
  const int n0   = blockIdx.x * M_TILE;
  const int tid  = threadIdx.x;
  const int wave = tid >> 5;
  const int lane = tid & 31;
  const int lhalf = lane >> 4;   // 0: lanes 0-15, 1: lanes 16-31
  const int l16   = lane & 15;

  // Async staging coords: thread copies rows (so) and (so+64), 8 elems at st8.
  const int so  = tid >> 2;          // 0..63
  const int st8 = (tid & 3) * 8;     // 0,8,16,24

  int cts[K_BLOCKS];
  #pragma unroll
  for (int k = 0; k < K_BLOCKS; ++k) cts[k] = col_idx[r * K_BLOCKS + k];

  v8f acc[8];
  #pragma unroll
  for (int i = 0; i < 8; ++i) {
    #pragma unroll
    for (int j = 0; j < 8; ++j) acc[i][j] = 0.0f;
  }

  auto issueW = [&](int c, int buf) {
    const int k  = c >> 2;
    const int kc = c & 3;
    const size_t gbase = (size_t)(r * K_BLOCKS + k) * TS * TS + (size_t)kc * KC + st8;
    const unsigned lh0 = (unsigned)(size_t)&sWhi[buf][ so       * LDS_STRIDE + st8];
    const unsigned lh1 = (unsigned)(size_t)&sWhi[buf][(so + 64) * LDS_STRIDE + st8];
    const unsigned ll0 = (unsigned)(size_t)&sWlo[buf][ so       * LDS_STRIDE + st8];
    const unsigned ll1 = (unsigned)(size_t)&sWlo[buf][(so + 64) * LDS_STRIDE + st8];
    asyncCopyB128(lh0, whi + gbase + (size_t)so * TS);
    asyncCopyB128(lh1, whi + gbase + (size_t)(so + 64) * TS);
    asyncCopyB128(ll0, wlo + gbase + (size_t)so * TS);
    asyncCopyB128(ll1, wlo + gbase + (size_t)(so + 64) * TS);
  };

  issueW(0, 0);

  const int  atb = lhalf * 8;    // A-fragment K sub-offset
  const int  btb = lhalf * 8;    // B-fragment K sub-offset

  #pragma unroll 1
  for (int c = 0; c < NCHUNK; ++c) {
    const int buf = c & 1;
    if (c + 1 < NCHUNK) {
      issueW(c + 1, buf ^ 1);
      WAIT_ASYNCCNT_4();         // chunk c's 4 ops have landed
    } else {
      WAIT_ASYNCCNT_0();
    }
    __syncthreads();

    const int k  = c >> 2;
    const int kc = c & 3;
    const int ct = cts[k];

    // ---- A fragments: direct global b128 loads (row-contiguous layout) ----
    const size_t arow  = (size_t)(n0 + wave * 16 + l16);
    const size_t abase = arow * C_IN + (size_t)ct * TS + (size_t)kc * KC + atb;
    Frag ahi, alo;
    ahi.q[0] = *(const uint4*)(xhi + abase);
    ahi.q[1] = *(const uint4*)(xhi + abase + 16);
    alo.q[0] = *(const uint4*)(xlo + abase);
    alo.q[1] = *(const uint4*)(xlo + abase + 16);

    // ---- B fragments from LDS + 3-term bf16 WMMA ----
    #pragma unroll
    for (int nt = 0; nt < 8; ++nt) {
      const int ocol = nt * 16 + l16;
      Frag bhi, blo;
      bhi.q[0] = *(const uint4*)&sWhi[buf][ocol * LDS_STRIDE + btb];
      bhi.q[1] = *(const uint4*)&sWhi[buf][ocol * LDS_STRIDE + btb + 16];
      blo.q[0] = *(const uint4*)&sWlo[buf][ocol * LDS_STRIDE + btb];
      blo.q[1] = *(const uint4*)&sWlo[buf][ocol * LDS_STRIDE + btb + 16];

      acc[nt] = __builtin_amdgcn_wmma_f32_16x16x32_bf16(
          false, ahi.v, false, bhi.v, (short)0, acc[nt], false, false);
      acc[nt] = __builtin_amdgcn_wmma_f32_16x16x32_bf16(
          false, ahi.v, false, blo.v, (short)0, acc[nt], false, false);
      acc[nt] = __builtin_amdgcn_wmma_f32_16x16x32_bf16(
          false, alo.v, false, bhi.v, (short)0, acc[nt], false, false);
    }
    __syncthreads();   // done reading buf before it is overwritten again
  }

  // ---- epilogue: C/D 16x16 f32 layout (vgpr i: M = i + 8*lhalf, N = l16) ----
  const int mbase = wave * 16 + lhalf * 8;
  float* ybase = y + (size_t)(n0 + mbase) * C_OUT + r * TS + l16;
  #pragma unroll
  for (int nt = 0; nt < 8; ++nt) {
    #pragma unroll
    for (int i = 0; i < 8; ++i) {
      ybase[(size_t)i * C_OUT + nt * 16] = acc[nt][i];
    }
  }
}

// ---------------------------------------------------------------------------
// Fallback (no workspace): self-contained kernel from round 1.
// ---------------------------------------------------------------------------
__global__ __launch_bounds__(256)
void bsr_wmma_bf16x3_kernel(const float* __restrict__ x,
                            const float* __restrict__ blocks,
                            const int*   __restrict__ col_idx,
                            float*       __restrict__ y)
{
  __shared__ __attribute__((aligned(16))) unsigned short sXhi[M_TILE * LDS_STRIDE];
  __shared__ __attribute__((aligned(16))) unsigned short sXlo[M_TILE * LDS_STRIDE];
  __shared__ __attribute__((aligned(16))) unsigned short sWhi[TS * LDS_STRIDE];
  __shared__ __attribute__((aligned(16))) unsigned short sWlo[TS * LDS_STRIDE];

  const int r    = blockIdx.y;
  const int n0   = blockIdx.x * M_TILE;
  const int tid  = threadIdx.x;
  const int wave = tid >> 5;
  const int lane = tid & 31;
  const int lhalf = lane >> 4;
  const int l16   = lane & 15;

  const int srow = tid >> 1;
  const int st   = (tid & 1) * 16;

  v8f acc[8];
  #pragma unroll
  for (int i = 0; i < 8; ++i) {
    #pragma unroll
    for (int j = 0; j < 8; ++j) acc[i][j] = 0.0f;
  }

  for (int k = 0; k < K_BLOCKS; ++k) {
    const int ct = col_idx[r * K_BLOCKS + k];
    const float* xsrc = x + (size_t)n0 * C_IN + (size_t)ct * TS;
    const float* wsrc = blocks + (size_t)(r * K_BLOCKS + k) * TS * TS;

    #pragma unroll 1
    for (int kc = 0; kc < TS / KC; ++kc) {
      {
        const float* px = xsrc + (size_t)srow * C_IN + kc * KC + st;
        const float* pw = wsrc + (size_t)srow * TS  + kc * KC + st;
        unsigned short xh[16], xl[16], wh[16], wl[16];
        #pragma unroll
        for (int i = 0; i < 4; ++i) {
          float4 vx = ((const float4*)px)[i];
          float4 vw = ((const float4*)pw)[i];
          splitBf16(vx.x, xh[4*i+0], xl[4*i+0]);
          splitBf16(vx.y, xh[4*i+1], xl[4*i+1]);
          splitBf16(vx.z, xh[4*i+2], xl[4*i+2]);
          splitBf16(vx.w, xh[4*i+3], xl[4*i+3]);
          splitBf16(vw.x, wh[4*i+0], wl[4*i+0]);
          splitBf16(vw.y, wh[4*i+1], wl[4*i+1]);
          splitBf16(vw.z, wh[4*i+2], wl[4*i+2]);
          splitBf16(vw.w, wh[4*i+3], wl[4*i+3]);
        }
        const int dbase = srow * LDS_STRIDE + st;
        #pragma unroll
        for (int h = 0; h < 2; ++h) {
          *(uint4*)&sXhi[dbase + 8*h] =
              make_uint4(pk2(xh[8*h+0], xh[8*h+1]), pk2(xh[8*h+2], xh[8*h+3]),
                         pk2(xh[8*h+4], xh[8*h+5]), pk2(xh[8*h+6], xh[8*h+7]));
          *(uint4*)&sXlo[dbase + 8*h] =
              make_uint4(pk2(xl[8*h+0], xl[8*h+1]), pk2(xl[8*h+2], xl[8*h+3]),
                         pk2(xl[8*h+4], xl[8*h+5]), pk2(xl[8*h+6], xl[8*h+7]));
          *(uint4*)&sWhi[dbase + 8*h] =
              make_uint4(pk2(wh[8*h+0], wh[8*h+1]), pk2(wh[8*h+2], wh[8*h+3]),
                         pk2(wh[8*h+4], wh[8*h+5]), pk2(wh[8*h+6], wh[8*h+7]));
          *(uint4*)&sWlo[dbase + 8*h] =
              make_uint4(pk2(wl[8*h+0], wl[8*h+1]), pk2(wl[8*h+2], wl[8*h+3]),
                         pk2(wl[8*h+4], wl[8*h+5]), pk2(wl[8*h+6], wl[8*h+7]));
        }
      }
      __syncthreads();

      const int arow = wave * 16 + l16;
      const int atb  = lhalf * 8;
      Frag ahi, alo;
      ahi.q[0] = *(const uint4*)&sXhi[arow * LDS_STRIDE + atb];
      ahi.q[1] = *(const uint4*)&sXhi[arow * LDS_STRIDE + 16 + atb];
      alo.q[0] = *(const uint4*)&sXlo[arow * LDS_STRIDE + atb];
      alo.q[1] = *(const uint4*)&sXlo[arow * LDS_STRIDE + 16 + atb];

      const int btb = lhalf * 8;
      #pragma unroll
      for (int nt = 0; nt < 8; ++nt) {
        const int ocol = nt * 16 + l16;
        Frag bhi, blo;
        bhi.q[0] = *(const uint4*)&sWhi[ocol * LDS_STRIDE + btb];
        bhi.q[1] = *(const uint4*)&sWhi[ocol * LDS_STRIDE + btb + 16];
        blo.q[0] = *(const uint4*)&sWlo[ocol * LDS_STRIDE + btb];
        blo.q[1] = *(const uint4*)&sWlo[ocol * LDS_STRIDE + btb + 16];

        acc[nt] = __builtin_amdgcn_wmma_f32_16x16x32_bf16(
            false, ahi.v, false, bhi.v, (short)0, acc[nt], false, false);
        acc[nt] = __builtin_amdgcn_wmma_f32_16x16x32_bf16(
            false, ahi.v, false, blo.v, (short)0, acc[nt], false, false);
        acc[nt] = __builtin_amdgcn_wmma_f32_16x16x32_bf16(
            false, alo.v, false, bhi.v, (short)0, acc[nt], false, false);
      }
      __syncthreads();
    }
  }

  const int mbase = wave * 16 + lhalf * 8;
  float* ybase = y + (size_t)(n0 + mbase) * C_OUT + r * TS + l16;
  #pragma unroll
  for (int nt = 0; nt < 8; ++nt) {
    #pragma unroll
    for (int i = 0; i < 8; ++i) {
      ybase[(size_t)i * C_OUT + nt * 16] = acc[nt][i];
    }
  }
}

extern "C" void kernel_launch(void* const* d_in, const int* in_sizes, int n_in,
                              void* d_out, int out_size, void* d_ws, size_t ws_size,
                              hipStream_t stream) {
  (void)in_sizes; (void)n_in; (void)out_size;
  const float* x      = (const float*)d_in[0];
  const float* blocks = (const float*)d_in[1];
  const int*   colidx = (const int*)d_in[2];
  float*       y      = (float*)d_out;

  dim3 grid(N_ROWS / M_TILE, R_BLOCKS);
  dim3 block(256);

  const size_t need = (XN + WN) * 2 * 2;   // hi+lo bf16 planes = 83,886,080 B
  if (ws_size >= need) {
    unsigned short* xhi = (unsigned short*)d_ws;
    unsigned short* xlo = xhi + XN;
    unsigned short* whi = xlo + XN;
    unsigned short* wlo = whi + WN;
    const size_t groups = (XN + WN) / 8;          // 8 elems per thread
    convert_split_kernel<<<dim3((unsigned)(groups / 256)), block, 0, stream>>>(
        x, blocks, xhi, xlo, whi, wlo);
    bsr_wmma_ws_kernel<<<grid, block, 0, stream>>>(xhi, xlo, whi, wlo, colidx, y);
  } else {
    bsr_wmma_bf16x3_kernel<<<grid, block, 0, stream>>>(x, blocks, colidx, y);
  }
}
